// NodeBondNet_12017318494548
// MI455X (gfx1250) — matrix operand order repaired and checked by reading, not compiled, not run on hardware
//
#include <hip/hip_runtime.h>
#include <cstdint>
#include <cstddef>

// ---------------- problem constants ----------------
#define N_NODES   16384
#define N_GRAPHS  256
#define M_PER_G   64
#define KNN       24
#define N_EDGES   (N_NODES * KNN)   // 393216
#define N_BONDS   32768
#define NODE_DIM  256
#define BOND_DIM  128
#define HID       256
#define N_GAUSS   20

// ---------------- feature detection ----------------
#if defined(__has_builtin)
#  if __has_builtin(__builtin_amdgcn_tensor_load_to_lds) && __has_builtin(__builtin_amdgcn_s_wait_tensorcnt)
#    define HAVE_TDM 1
#  endif
#endif
#ifndef HAVE_TDM
#  define HAVE_TDM 0
#  warning "CDNA5 TDM builtins not available on this toolchain; falling back to manual LDS staging"
#endif

// ---------------- WMMA types / helpers ----------------
typedef __attribute__((ext_vector_type(16))) __bf16 bf16x16;
typedef __attribute__((ext_vector_type(8)))  float  f32x8;
typedef __attribute__((ext_vector_type(4)))  unsigned int u32x4;
typedef __attribute__((ext_vector_type(8)))  int  i32x8;
typedef __attribute__((ext_vector_type(4)))  int  i32x4;

struct bf16x4s { __bf16 x, y, z, w; };

__device__ __forceinline__ __bf16 f2bf(float f) {
  union { float f; uint32_t u; } x; x.f = f;
  uint32_t r = (x.u + 0x7fffu + ((x.u >> 16) & 1u)) >> 16;
  union { unsigned short s; __bf16 b; } y; y.s = (unsigned short)r;
  return y.b;
}
__device__ __forceinline__ float bf2f(__bf16 b) {
  union { unsigned short s; __bf16 b; } y; y.b = b;
  union { uint32_t u; float f; } x; x.u = ((uint32_t)y.s) << 16;
  return x.f;
}

union FragCast { bf16x16 v; uint4 q[2]; };

// A-fragment: 16x32 bf16 tile, row-major with `stride` (elements, multiple of 8).
// lanes 0-15: row=lane, K {0..7,16..23}; lanes 16-31: row=lane-16, K {8..15,24..31}
__device__ __forceinline__ bf16x16 load_frag_a(const __bf16* tile, int stride) {
  int lane = threadIdx.x & 31;
  int r = lane & 15;
  int ko = (lane >> 4) << 3;                // 0 or 8
  const __bf16* p = tile + r * stride + ko;
  FragCast fc;
  fc.q[0] = *(const uint4*)(p);             // K base..base+7
  fc.q[1] = *(const uint4*)(p + 16);        // K base+16..base+23
  return fc.v;
}
// B-fragment from W stored [N][K] row-major: lane<16 -> col n=lane, K {0..7,16..23};
// lane>=16 -> col n=lane-16, K {8..15,24..31}.
__device__ __forceinline__ bf16x16 load_frag_b(const __bf16* tile, int stride) {
  int lane = threadIdx.x & 31;
  int n = lane & 15;
  const __bf16* p = tile + n * stride + ((lane >> 4) << 3); // 0 or 8
  FragCast fc;
  fc.q[0] = *(const uint4*)(p);
  fc.q[1] = *(const uint4*)(p + 16);
  return fc.v;
}
__device__ __forceinline__ f32x8 wmma32(bf16x16 a, bf16x16 b, f32x8 c) {
  return __builtin_amdgcn_wmma_f32_16x16x32_bf16(false, a, false, b, (short)0, c, false, false);
}

// Issue one 1-D TDM copy of `bytes` (multiple of 8) from global `src` to LDS offset
// `ldsAddr`. Must be executed by a single wave; follow with s_wait_tensorcnt + barrier.
#if HAVE_TDM
__device__ __forceinline__ void tdm_copy_1d(const void* src, uint32_t ldsAddr, uint32_t bytes) {
  uint64_t ga = (uint64_t)(uintptr_t)src;
  uint32_t nb8 = bytes >> 3;                     // 8-byte units (<= 4096 here)
  u32x4 g0;
  g0[0] = 1u;                                    // count=1, user descriptor
  g0[1] = ldsAddr;                               // lds_addr (bytes)
  g0[2] = (uint32_t)ga;                          // global_addr[31:0]
  g0[3] = ((uint32_t)(ga >> 32) & 0x01FFFFFFu) | 0x80000000u;  // addr[56:32] | type=2
  i32x8 g1;
  g1[0] = (int)(3u << 16);                       // data_size=3 (8B), no multicast/pad/iter
  g1[1] = (int)((nb8 & 0xFFFFu) << 16);          // tensor_dim0[15:0]
  g1[2] = (int)(((nb8 >> 16) & 0xFFFFu) | (1u << 16)); // tensor_dim0[31:16] | tensor_dim1=1 lo
  g1[3] = (int)(nb8 << 16);                      // tensor_dim1 hi=0 | tile_dim0 = nb8
  g1[4] = 0;                                     // tile_dim1=0 (1-D), tile_dim2=0
  g1[5] = (int)nb8;                              // tensor_dim0_stride lo
  g1[6] = 0;
  g1[7] = 0;
  i32x4 z4 = {0, 0, 0, 0};
#if __clang_major__ >= 23
  i32x8 z8 = {0, 0, 0, 0, 0, 0, 0, 0};
  __builtin_amdgcn_tensor_load_to_lds(g0, g1, z4, z4, z8, 0);
#else
  __builtin_amdgcn_tensor_load_to_lds(g0, g1, z4, z4, 0);
#endif
}
#endif

// ---------------- epilogue codes ----------------
#define EP_F32          0
#define EP_BF16         1
#define EP_LNRELU_BF16  2
#define EP_F32_RESID    3
#define EP_SCATTER_ADD  4

// ================= generic 64-row WMMA GEMM =================
// Y[rowBase..rowBase+63][0..N) = ep( A(row) (optionally * Amul[mulIdx[row]]) @ W^T + bias )
// A: bf16 [M][K]; W: bf16 [N][K] row-major; K,N in {128,256}.
// Each wave owns a column tile and all 4 row tiles: one B-fragment feeds 4 WMMAs.
__global__ __launch_bounds__(256) void gemm64(
    const __bf16* __restrict__ A, const __bf16* __restrict__ Amul,
    const int* __restrict__ mulIdx,
    const __bf16* __restrict__ W, const float* __restrict__ bias,
    int K, int N, int ep,
    float* __restrict__ outF, __bf16* __restrict__ outB,
    const float* __restrict__ lnG, const float* __restrict__ lnBt,
    const float* __restrict__ resid, const int* __restrict__ scatterIdx)
{
  __shared__ __bf16 As[64 * 256];
  __shared__ float  Ds[64 * 256];
  const int tid = threadIdx.x;
  const int rowBase = blockIdx.x * 64;

  __builtin_prefetch(W, 0, 1);

  // ---- stage A tile into LDS as bf16 ----
  bool staged = false;
#if HAVE_TDM
  if (Amul == nullptr) {
    // contiguous 64 x K bf16 rows: single async tensor DMA into LDS
    if (tid < 32) {
      tdm_copy_1d(A + (size_t)rowBase * K, (uint32_t)(uintptr_t)(void*)As,
                  (uint32_t)(64 * K * 2));
      __builtin_amdgcn_s_wait_tensorcnt(0);
    }
    staged = true;
  }
#endif
  if (!staged) {
    for (int t = tid; t < 64 * K; t += 256) {
      int r = t / K, k = t - r * K;
      size_t src = (size_t)(rowBase + r) * K + k;
      float v = bf2f(A[src]);
      if (Amul) {
        int mr = mulIdx ? mulIdx[rowBase + r] : (rowBase + r);
        v *= bf2f(Amul[(size_t)mr * K + k]);
      }
      As[r * K + k] = f2bf(v);
    }
  }
  __syncthreads();

  const int wave = tid >> 5;
  const int lane = tid & 31;
  const int CT = N >> 4;                    // col tiles (8 or 16)
  for (int ct = wave; ct < CT; ct += 8) {
    f32x8 acc[4] = {{}, {}, {}, {}};
    for (int kk = 0; kk < (K >> 5); ++kk) {
      bf16x16 b = load_frag_b(W + (ct * 16) * K + kk * 32, K);
#pragma unroll
      for (int mt = 0; mt < 4; ++mt)
        acc[mt] = wmma32(load_frag_a(As + (mt * 16) * K + kk * 32, K), b, acc[mt]);
    }
    int n = ct * 16 + (lane & 15);
    float bs = bias ? bias[n] : 0.f;
#pragma unroll
    for (int mt = 0; mt < 4; ++mt) {
      int m = mt * 16 + ((lane >> 4) << 3);
#pragma unroll
      for (int r = 0; r < 8; ++r) Ds[(m + r) * N + n] = acc[mt][r] + bs;
    }
  }
  __syncthreads();

  if (ep == EP_LNRELU_BF16) {
    if (tid < 64) {
      int r = tid;
      float s = 0.f, ss = 0.f;
      for (int j = 0; j < N; ++j) { float v = Ds[r * N + j]; s += v; ss += v * v; }
      float mu = s / (float)N;
      float var = ss / (float)N - mu * mu;
      float rs = rsqrtf(var + 1e-5f);
      for (int j = 0; j < N; ++j) {
        float v = (Ds[r * N + j] - mu) * rs * lnG[j] + lnBt[j];
        outB[(size_t)(rowBase + r) * N + j] = f2bf(v > 0.f ? v : 0.f);
      }
    }
  } else if (ep == EP_SCATTER_ADD) {
    for (int t = tid; t < 64 * N; t += 256) {
      int r = t / N, j = t - r * N;
      int dst = scatterIdx[rowBase + r];
      atomicAdd(outF + (size_t)dst * N + j, Ds[t]);
    }
  } else {
    const int NV = N >> 2;                  // float4 chunks per row
    for (int t = tid; t < 64 * NV; t += 256) {
      int r = t / NV, j4 = (t - r * NV) << 2;
      size_t g = (size_t)(rowBase + r) * N + j4;
      float4 v = *(const float4*)&Ds[r * N + j4];
      if (ep == EP_F32_RESID) {
        float4 rr = *(const float4*)&resid[g];
        v.x += rr.x; v.y += rr.y; v.z += rr.z; v.w += rr.w;
      }
      if (ep == EP_BF16) {
        bf16x4s o; o.x = f2bf(v.x); o.y = f2bf(v.y); o.z = f2bf(v.z); o.w = f2bf(v.w);
        *(bf16x4s*)&outB[g] = o;
      } else {
        *(float4*)&outF[g] = v;
      }
    }
  }
}

// ================= fused kNN edge kernel =================
#define NPB 2
#define EPB (NPB * KNN)   // 48

// One wave per column tile (2 per wave), 3 row tiles per wave: B-frag reused 3x.
// KT = K/32, AST = A row stride (elements).
template <int KT, int AST>
__device__ __forceinline__ void edge_gemm(const __bf16* As0, const __bf16* __restrict__ W,
                                          const float* __restrict__ bias,
                                          float (&stage)[EPB][256]) {
  const int tid = threadIdx.x;
  const int wave = tid >> 5, lane = tid & 31;
  const int WK = KT * 32;
  for (int ct = wave; ct < 16; ct += 8) {
    f32x8 acc[3] = {{}, {}, {}};
    for (int kk = 0; kk < KT; ++kk) {
      bf16x16 b = load_frag_b(W + (ct * 16) * WK + kk * 32, WK);
#pragma unroll
      for (int mt = 0; mt < 3; ++mt)
        acc[mt] = wmma32(load_frag_a(As0 + (mt * 16) * AST + kk * 32, AST), b, acc[mt]);
    }
    int n = ct * 16 + (lane & 15);
    float bs = bias[n];
#pragma unroll
    for (int mt = 0; mt < 3; ++mt) {
      int m = mt * 16 + ((lane >> 4) << 3);
#pragma unroll
      for (int r = 0; r < 8; ++r) stage[m + r][n] = acc[mt][r] + bs;
    }
  }
}

// Per block: 2 nodes = 48 edges. Chain fully in LDS:
//   eattr = edge_emb([gauss(20), onehot(src),onehot(dst)])
//   h1    = relu(LN(eattr @ W1^T + b1))
//   e2    = h1 @ W2^T + b2
//   prod  = e2 * nn[col]
//   msg   = prod @ Wm^T + bm
//   aggr[node] += sum_{24 edges} msg        (exclusive per block; no atomics)
__global__ __launch_bounds__(256) void edge_block_kernel(
    const int* __restrict__ col, const float* __restrict__ dist,
    const int* __restrict__ tc,
    const float* __restrict__ We, const float* __restrict__ be,    // edge_emb [128,26],[128]
    const __bf16* __restrict__ W1, const float* __restrict__ b1,   // edge_net l1 [256,128]
    const float* __restrict__ g1, const float* __restrict__ be1,   // edge_net ln
    const __bf16* __restrict__ W2, const float* __restrict__ b2,   // edge_net l2 [256,256]
    const __bf16* __restrict__ Wm, const float* __restrict__ bm,   // msg [256,256]
    const __bf16* __restrict__ nn,                                  // node_net out [N,256] bf16
    float* __restrict__ aggr)                                       // [N,256] pre-filled with cent
{
  __shared__ float  feat[EPB][28];
  __shared__ int    colL[EPB];
  __shared__ __bf16 eattr[EPB][136];   // 128 used, stride mult of 8
  __shared__ float  stage[EPB][256];
  __shared__ __bf16 hbuf[EPB][264];    // 256 used

  const int tid = threadIdx.x;
  const int nodeBase = blockIdx.x * NPB;
  const int ebase = nodeBase * KNN;

  __builtin_prefetch(W1, 0, 1);
  __builtin_prefetch(W2, 0, 1);
  __builtin_prefetch(Wm, 0, 1);

  // phase 1: per-edge raw features
  if (tid < EPB) {
    int e = ebase + tid;
    int c = col[e];
    colL[tid] = c;
    float d = dist[e];
    const float delta = 10.0f / 19.0f;           // linspace(0,10,20) step
    const float coeff = -0.5f / (delta * delta);
#pragma unroll
    for (int j = 0; j < N_GAUSS; ++j) {
      float t = d - delta * (float)j;
      feat[tid][j] = __expf(coeff * t * t);
    }
    int srcT = tc[nodeBase + tid / KNN];
    int dstT = tc[c];
#pragma unroll
    for (int j = 0; j < 3; ++j) {
      feat[tid][20 + j] = (srcT == j) ? 1.f : 0.f;
      feat[tid][23 + j] = (dstT == j) ? 1.f : 0.f;
    }
  }
  __syncthreads();

  // phase 2: eattr = feat @ We^T + be (26 -> 128), VALU (tiny)
  for (int o = tid; o < EPB * 128; o += 256) {
    int e = o >> 7, j = o & 127;
    float acc = be[j];
    const float* wr = We + j * 26;
#pragma unroll
    for (int q = 0; q < 26; ++q) acc += wr[q] * feat[e][q];
    eattr[e][j] = f2bf(acc);
  }
  __syncthreads();

  // phase 3: GEMM1 (K=128) -> stage + b1
  edge_gemm<4, 136>(&eattr[0][0], W1, b1, stage);
  __syncthreads();

  // phase 4: LayerNorm + ReLU -> hbuf (bf16)
  if (tid < EPB) {
    float s = 0.f, ss = 0.f;
    for (int j = 0; j < 256; ++j) { float v = stage[tid][j]; s += v; ss += v * v; }
    float mu = s * (1.f / 256.f);
    float var = ss * (1.f / 256.f) - mu * mu;
    float rs = rsqrtf(var + 1e-5f);
    for (int j = 0; j < 256; ++j) {
      float v = (stage[tid][j] - mu) * rs * g1[j] + be1[j];
      hbuf[tid][j] = f2bf(v > 0.f ? v : 0.f);
    }
  }
  __syncthreads();

  // phase 5: GEMM2 (K=256) -> stage + b2
  edge_gemm<8, 264>(&hbuf[0][0], W2, b2, stage);
  __syncthreads();

  // phase 6: multiply by node_net[col]
  for (int o = tid; o < EPB * 256; o += 256) {
    int e = o >> 8, j = o & 255;
    float v = stage[e][j] * bf2f(nn[(size_t)colL[e] * 256 + j]);
    hbuf[e][j] = f2bf(v);
  }
  __syncthreads();

  // phase 7: GEMM3 msg (K=256) -> stage + bm
  edge_gemm<8, 264>(&hbuf[0][0], Wm, bm, stage);
  __syncthreads();

  // phase 8: reduce groups of 24 and accumulate into aggr (exclusive rows)
  for (int o = tid; o < NPB * 256; o += 256) {
    int nl = o >> 8, j = o & 255;
    float s = 0.f;
#pragma unroll
    for (int t = 0; t < KNN; ++t) s += stage[nl * KNN + t][j];
    size_t g = (size_t)(nodeBase + nl) * 256 + j;
    aggr[g] += s;
  }
}

// ================= small kernels =================
__global__ __launch_bounds__(64) void knn_kernel(const float* __restrict__ pos,
                                                 int* __restrict__ col,
                                                 float* __restrict__ dist) {
  __shared__ float px[M_PER_G], py[M_PER_G], pz[M_PER_G];
  const int g = blockIdx.x, t = threadIdx.x;
  const int nb = g * M_PER_G;
  px[t] = pos[(size_t)(nb + t) * 3 + 0];
  py[t] = pos[(size_t)(nb + t) * 3 + 1];
  pz[t] = pos[(size_t)(nb + t) * 3 + 2];
  __syncthreads();
  float d2[M_PER_G];
  for (int j = 0; j < M_PER_G; ++j) {
    float dx = px[t] - px[j], dy = py[t] - py[j], dz = pz[t] - pz[j];
    d2[j] = dx * dx + dy * dy + dz * dz;
  }
  d2[t] = 1e30f;  // exclude self
  unsigned long long used = 0ull;
  for (int k = 0; k < KNN; ++k) {
    float best = 1e29f; int bi = 0;
    for (int j = 0; j < M_PER_G; ++j) {
      if (!((used >> j) & 1ull) && d2[j] < best) { best = d2[j]; bi = j; }
    }
    used |= (1ull << bi);
    int e = (nb + t) * KNN + k;
    col[e] = nb + bi;
    dist[e] = sqrtf(best);
  }
}

__global__ __launch_bounds__(256) void typecode_kernel(const unsigned char* __restrict__ is_mol,
                                                       const unsigned char* __restrict__ is_frag,
                                                       int* __restrict__ tc, int n) {
  int i = blockIdx.x * 256 + threadIdx.x;
  if (i < n) tc[i] = (is_mol[i] ? 1 : 0) + (is_frag[i] ? 2 : 0);
}

// n must be a multiple of 4 and pointers 16B/8B aligned (true for all call sites)
__global__ __launch_bounds__(256) void cvt_bf16_kernel(const float* __restrict__ x,
                                                       __bf16* __restrict__ y, int n) {
  int i = (blockIdx.x * 256 + threadIdx.x) << 2;
  if (i < n) {
    float4 v = *(const float4*)(x + i);
    bf16x4s o; o.x = f2bf(v.x); o.y = f2bf(v.y); o.z = f2bf(v.z); o.w = f2bf(v.w);
    *(bf16x4s*)(y + i) = o;
  }
}

__global__ __launch_bounds__(256) void zero_kernel(float* __restrict__ x, int n) {
  int i = blockIdx.x * 256 + threadIdx.x;
  if (i < n) x[i] = 0.f;
}

__global__ __launch_bounds__(256) void add2_bf16_kernel(const float* __restrict__ a,
                                                        const float* __restrict__ b,
                                                        __bf16* __restrict__ y, int n) {
  int i = (blockIdx.x * 256 + threadIdx.x) << 2;
  if (i < n) {
    float4 va = *(const float4*)(a + i);
    float4 vb = *(const float4*)(b + i);
    bf16x4s o;
    o.x = f2bf(va.x + vb.x); o.y = f2bf(va.y + vb.y);
    o.z = f2bf(va.z + vb.z); o.w = f2bf(va.w + vb.w);
    *(bf16x4s*)(y + i) = o;
  }
}

// one row per block, blockDim == N (128 or 256, power of 2)
__global__ void row_ln_relu_kernel(const float* __restrict__ x, const float* __restrict__ g,
                                   const float* __restrict__ b, __bf16* __restrict__ y, int N) {
  __shared__ float red[256];
  const int r = blockIdx.x, j = threadIdx.x;
  float v = x[(size_t)r * N + j];
  red[j] = v;
  __syncthreads();
  for (int s = N >> 1; s > 0; s >>= 1) { if (j < s) red[j] += red[j + s]; __syncthreads(); }
  float mu = red[0] / (float)N;
  __syncthreads();
  float d = v - mu;
  red[j] = d * d;
  __syncthreads();
  for (int s = N >> 1; s > 0; s >>= 1) { if (j < s) red[j] += red[j + s]; __syncthreads(); }
  float rs = rsqrtf(red[0] / (float)N + 1e-5f);
  float o = d * rs * g[j] + b[j];
  y[(size_t)r * N + j] = f2bf(o > 0.f ? o : 0.f);
}

// hb[e][:] += mL[left[e]] + mR[right[e]] + nL[left[e]] + nR[right[e]]
__global__ __launch_bounds__(256) void bond_combine_kernel(
    float* __restrict__ hb, const float* __restrict__ mL, const float* __restrict__ mR,
    const float* __restrict__ nL, const float* __restrict__ nR,
    const int* __restrict__ left, const int* __restrict__ right) {
  int t = blockIdx.x * 256 + threadIdx.x;   // over N_BONDS*128
  int e = t >> 7, j = t & 127;
  int l = left[e], r = right[e];
  hb[t] += mL[(size_t)l * 128 + j] + mR[(size_t)r * 128 + j]
         + nL[(size_t)l * 128 + j] + nR[(size_t)r * 128 + j];
}

// ================= host orchestration =================
// Param leaf indexing (recursive insertion-order flatten, None biases dropped):
// 0..5: h_node, pos_node, h_bond, bond_index, is_mol, is_frag
// 6: edge_emb.w  7: edge_emb.b  8: lin_node.w  9: lin_node.b
// block b @ 10 + 66*b:
//   node_edge @+0  (20): nn.l1.w,b, nn.ln.g,b, nn.l2.w,b, en.l1.w,b, en.ln.g,b,
//                        en.l2.w,b, msg.w,b, cent.w,b, ln.g,b, out.w,b
//   node_bond @+20 (20): same layout
//   bond      @+40 (26): fl.bond.w, fl.node.w, fl.l1.w,b, fl.ln.g,b, fl.l2.w,b,
//                        fr.(8), node_l.w,b, node_r.w,b, self.w,b, ln.g,b, out.w,b

extern "C" void kernel_launch(void* const* d_in, const int* in_sizes, int n_in,
                              void* d_out, int out_size, void* d_ws, size_t ws_size,
                              hipStream_t stream) {
  (void)in_sizes; (void)n_in; (void)out_size; (void)ws_size;

  const float* h_node0 = (const float*)d_in[0];
  const float* pos     = (const float*)d_in[1];
  const float* h_bond0 = (const float*)d_in[2];
  const int*   bondIdx = (const int*)d_in[3];
  const unsigned char* is_mol  = (const unsigned char*)d_in[4];
  const unsigned char* is_frag = (const unsigned char*)d_in[5];
  const int* leftI  = bondIdx;
  const int* rightI = bondIdx + N_BONDS;

  float* outNode = (float*)d_out;
  float* outBond = (float*)d_out + (size_t)N_NODES * NODE_DIM;

  char* wsBase = (char*)d_ws;
  size_t off = 0;
  auto alloc = [&](size_t bytes) -> void* {
    void* p = wsBase + off;
    off = (off + bytes + 255) & ~(size_t)255;
    return p;
  };

  // persistent scratch
  int*    col   = (int*)  alloc((size_t)N_EDGES * 4);
  float*  dist  = (float*)alloc((size_t)N_EDGES * 4);
  int*    tc    = (int*)  alloc((size_t)N_NODES * 4);
  float*  hnode1 = (float*)alloc((size_t)N_NODES * NODE_DIM * 4);
  float*  hbond1 = (float*)alloc((size_t)N_BONDS * BOND_DIM * 4);
  __bf16* hnB  = (__bf16*)alloc((size_t)N_NODES * NODE_DIM * 2);
  __bf16* hbB  = (__bf16*)alloc((size_t)N_BONDS * BOND_DIM * 2);
  __bf16* tmpB = (__bf16*)alloc((size_t)N_NODES * HID * 2);
  __bf16* nnE  = (__bf16*)alloc((size_t)N_NODES * HID * 2);
  __bf16* nnBf = (__bf16*)alloc((size_t)N_NODES * HID * 2);
  float*  aggrE = (float*)alloc((size_t)N_NODES * HID * 4);
  float*  aggrB = (float*)alloc((size_t)N_NODES * HID * 4);
  __bf16* tbuf = (__bf16*)alloc((size_t)N_NODES * HID * 2);
  float*  h_ne = (float*)alloc((size_t)N_NODES * NODE_DIM * 4);
  float*  h_nb = (float*)alloc((size_t)N_NODES * NODE_DIM * 4);
  __bf16* comb = (__bf16*)alloc((size_t)N_NODES * NODE_DIM * 2);
  // bond block scratch
  __bf16* blB  = (__bf16*)alloc((size_t)N_BONDS * 256 * 2);
  __bf16* nlB  = (__bf16*)alloc((size_t)N_NODES * 256 * 2);
  __bf16* ihB  = (__bf16*)alloc((size_t)N_BONDS * 256 * 2);
  float*  mLn  = (float*)alloc((size_t)N_NODES * 128 * 4);
  float*  mRn  = (float*)alloc((size_t)N_NODES * 128 * 4);
  float*  nLn  = (float*)alloc((size_t)N_NODES * 128 * 4);
  float*  nRn  = (float*)alloc((size_t)N_NODES * 128 * 4);
  float*  hbf  = (float*)alloc((size_t)N_BONDS * 128 * 4);
  __bf16* hbr  = (__bf16*)alloc((size_t)N_BONDS * 128 * 2);
  __bf16* h1b  = (__bf16*)alloc((size_t)N_BONDS * 256 * 2);
  __bf16* e2b  = (__bf16*)alloc((size_t)N_BONDS * 256 * 2);

  auto wcvt = [&](int idx, size_t n) -> const __bf16* {
    __bf16* dst = (__bf16*)alloc(n * 2);
    cvt_bf16_kernel<<<(int)((n / 4 + 255) / 256), 256, 0, stream>>>((const float*)d_in[idx], dst,
                                                                    (int)n);
    return dst;
  };
  auto F = [&](int idx) -> const float* { return (const float*)d_in[idx]; };

  auto gemm = [&](const __bf16* A, const __bf16* Amul, const int* mulIdx,
                  const __bf16* W, const float* bias, int M, int K, int N, int ep,
                  float* oF, __bf16* oB, const float* g, const float* b,
                  const float* resid, const int* scat) {
    gemm64<<<M / 64, 256, 0, stream>>>(A, Amul, mulIdx, W, bias, K, N, ep,
                                       oF, oB, g, b, resid, scat);
  };

  // ---- graph build (layer-independent) ----
  knn_kernel<<<N_GRAPHS, M_PER_G, 0, stream>>>(pos, col, dist);
  typecode_kernel<<<(N_NODES + 255) / 256, 256, 0, stream>>>(is_mol, is_frag, tc, N_NODES);

  const __bf16* WlinNode = wcvt(8, (size_t)256 * 256);
  const float*  blinNode = F(9);

  const float* hn = h_node0;
  const float* hb = h_bond0;

  for (int blk = 0; blk < 2; ++blk) {
    const int B  = 10 + 66 * blk;
    const int NE = B + 0, NB = B + 20, BD = B + 40;
    float* hnOut = (blk == 0) ? hnode1 : outNode;
    float* hbOut = (blk == 0) ? hbond1 : outBond;

    // bf16 copies of current activations
    cvt_bf16_kernel<<<(N_NODES * NODE_DIM / 4 + 255) / 256, 256, 0, stream>>>(
        hn, hnB, N_NODES * NODE_DIM);
    cvt_bf16_kernel<<<(N_BONDS * BOND_DIM / 4 + 255) / 256, 256, 0, stream>>>(
        hb, hbB, N_BONDS * BOND_DIM);

    // weight conversions for this layer
    const __bf16* neNN1 = wcvt(NE + 0, 256 * 256);
    const __bf16* neNN2 = wcvt(NE + 4, 256 * 256);
    const __bf16* neE1  = wcvt(NE + 6, 256 * 128);
    const __bf16* neE2  = wcvt(NE + 10, 256 * 256);
    const __bf16* neMsg = wcvt(NE + 12, 256 * 256);
    const __bf16* neCent= wcvt(NE + 14, 256 * 256);
    const __bf16* neOut = wcvt(NE + 18, 256 * 256);
    const __bf16* nbNN1 = wcvt(NB + 0, 256 * 256);
    const __bf16* nbNN2 = wcvt(NB + 4, 256 * 256);
    const __bf16* nbE1  = wcvt(NB + 6, 256 * 128);
    const __bf16* nbE2  = wcvt(NB + 10, 256 * 256);
    const __bf16* nbMsg = wcvt(NB + 12, 256 * 256);
    const __bf16* nbCent= wcvt(NB + 14, 256 * 256);
    const __bf16* nbOut = wcvt(NB + 18, 256 * 256);
    const __bf16* flBond = wcvt(BD + 0, 256 * 128);
    const __bf16* flNode = wcvt(BD + 1, 256 * 256);
    const __bf16* flL1   = wcvt(BD + 2, 256 * 256);
    const __bf16* flL2   = wcvt(BD + 6, 128 * 256);
    const __bf16* frBond = wcvt(BD + 8, 256 * 128);
    const __bf16* frNode = wcvt(BD + 9, 256 * 256);
    const __bf16* frL1   = wcvt(BD + 10, 256 * 256);
    const __bf16* frL2   = wcvt(BD + 14, 128 * 256);
    const __bf16* bNodeL = wcvt(BD + 16, 128 * 256);
    const __bf16* bNodeR = wcvt(BD + 18, 128 * 256);
    const __bf16* bSelf  = wcvt(BD + 20, 128 * 128);
    const __bf16* bOut   = wcvt(BD + 24, 128 * 128);

    // ---- node_net MLPs (per node, both node blocks) ----
    gemm(hnB, nullptr, nullptr, neNN1, F(NE + 1), N_NODES, 256, 256, EP_LNRELU_BF16,
         nullptr, tmpB, F(NE + 2), F(NE + 3), nullptr, nullptr);
    gemm(tmpB, nullptr, nullptr, neNN2, F(NE + 5), N_NODES, 256, 256, EP_BF16,
         nullptr, nnE, nullptr, nullptr, nullptr, nullptr);
    gemm(hnB, nullptr, nullptr, nbNN1, F(NB + 1), N_NODES, 256, 256, EP_LNRELU_BF16,
         nullptr, tmpB, F(NB + 2), F(NB + 3), nullptr, nullptr);
    gemm(tmpB, nullptr, nullptr, nbNN2, F(NB + 5), N_NODES, 256, 256, EP_BF16,
         nullptr, nnBf, nullptr, nullptr, nullptr, nullptr);

    // ---- cent linears pre-fill aggregation buffers ----
    gemm(hnB, nullptr, nullptr, neCent, F(NE + 15), N_NODES, 256, 256, EP_F32,
         aggrE, nullptr, nullptr, nullptr, nullptr, nullptr);
    gemm(hnB, nullptr, nullptr, nbCent, F(NB + 15), N_NODES, 256, 256, EP_F32,
         aggrB, nullptr, nullptr, nullptr, nullptr, nullptr);

    // ---- fused kNN edge path (adds into aggrE) ----
    edge_block_kernel<<<N_NODES / NPB, 256, 0, stream>>>(
        col, dist, tc, F(6), F(7),
        neE1, F(NE + 7), F(NE + 8), F(NE + 9),
        neE2, F(NE + 11), neMsg, F(NE + 13), nnE, aggrE);

    // ---- node_bond edge path (bond graph; atomics into aggrB) ----
    gemm(hbB, nullptr, nullptr, nbE1, F(NB + 7), N_BONDS, 128, 256, EP_LNRELU_BF16,
         nullptr, h1b, F(NB + 8), F(NB + 9), nullptr, nullptr);
    gemm(h1b, nullptr, nullptr, nbE2, F(NB + 11), N_BONDS, 256, 256, EP_BF16,
         nullptr, e2b, nullptr, nullptr, nullptr, nullptr);
    gemm(e2b, nnBf, rightI, nbMsg, F(NB + 13), N_BONDS, 256, 256, EP_SCATTER_ADD,
         aggrB, nullptr, nullptr, nullptr, nullptr, leftI);

    // ---- node-block epilogues -> h_ne, h_nb ----
    row_ln_relu_kernel<<<N_NODES, 256, 0, stream>>>(aggrE, F(NE + 16), F(NE + 17), tbuf, 256);
    gemm(tbuf, nullptr, nullptr, neOut, F(NE + 19), N_NODES, 256, 256, EP_F32,
         h_ne, nullptr, nullptr, nullptr, nullptr, nullptr);
    row_ln_relu_kernel<<<N_NODES, 256, 0, stream>>>(aggrB, F(NB + 16), F(NB + 17), tbuf, 256);
    gemm(tbuf, nullptr, nullptr, nbOut, F(NB + 19), N_NODES, 256, 256, EP_F32,
         h_nb, nullptr, nullptr, nullptr, nullptr, nullptr);

    // ---- bond block ----
    zero_kernel<<<(N_NODES * 128 + 255) / 256, 256, 0, stream>>>(mLn, N_NODES * 128);
    zero_kernel<<<(N_NODES * 128 + 255) / 256, 256, 0, stream>>>(mRn, N_NODES * 128);
    // ffn_l
    gemm(hbB, nullptr, nullptr, flBond, nullptr, N_BONDS, 128, 256, EP_BF16,
         nullptr, blB, nullptr, nullptr, nullptr, nullptr);
    gemm(hnB, nullptr, nullptr, flNode, nullptr, N_NODES, 256, 256, EP_BF16,
         nullptr, nlB, nullptr, nullptr, nullptr, nullptr);
    gemm(blB, nlB, leftI, flL1, F(BD + 3), N_BONDS, 256, 256, EP_LNRELU_BF16,
         nullptr, ihB, F(BD + 4), F(BD + 5), nullptr, nullptr);
    gemm(ihB, nullptr, nullptr, flL2, F(BD + 7), N_BONDS, 256, 128, EP_SCATTER_ADD,
         mLn, nullptr, nullptr, nullptr, nullptr, rightI);
    // ffn_r
    gemm(hbB, nullptr, nullptr, frBond, nullptr, N_BONDS, 128, 256, EP_BF16,
         nullptr, blB, nullptr, nullptr, nullptr, nullptr);
    gemm(hnB, nullptr, nullptr, frNode, nullptr, N_NODES, 256, 256, EP_BF16,
         nullptr, nlB, nullptr, nullptr, nullptr, nullptr);
    gemm(blB, nlB, rightI, frL1, F(BD + 11), N_BONDS, 256, 256, EP_LNRELU_BF16,
         nullptr, ihB, F(BD + 12), F(BD + 13), nullptr, nullptr);
    gemm(ihB, nullptr, nullptr, frL2, F(BD + 15), N_BONDS, 256, 128, EP_SCATTER_ADD,
         mRn, nullptr, nullptr, nullptr, nullptr, leftI);
    // node_l / node_r / self
    gemm(hnB, nullptr, nullptr, bNodeL, F(BD + 17), N_NODES, 256, 128, EP_F32,
         nLn, nullptr, nullptr, nullptr, nullptr, nullptr);
    gemm(hnB, nullptr, nullptr, bNodeR, F(BD + 19), N_NODES, 256, 128, EP_F32,
         nRn, nullptr, nullptr, nullptr, nullptr, nullptr);
    gemm(hbB, nullptr, nullptr, bSelf, F(BD + 21), N_BONDS, 128, 128, EP_F32,
         hbf, nullptr, nullptr, nullptr, nullptr, nullptr);
    bond_combine_kernel<<<(N_BONDS * 128) / 256, 256, 0, stream>>>(hbf, mLn, mRn, nLn, nRn,
                                                                   leftI, rightI);
    row_ln_relu_kernel<<<N_BONDS, 128, 0, stream>>>(hbf, F(BD + 22), F(BD + 23), hbr, 128);
    gemm(hbr, nullptr, nullptr, bOut, F(BD + 25), N_BONDS, 128, 128, EP_F32_RESID,
         hbOut, nullptr, nullptr, nullptr, hb, nullptr);

    // ---- node update: h_node += lin_node(h_ne + h_nb) ----
    add2_bf16_kernel<<<(N_NODES * NODE_DIM / 4 + 255) / 256, 256, 0, stream>>>(
        h_ne, h_nb, comb, N_NODES * NODE_DIM);
    gemm(comb, nullptr, nullptr, WlinNode, blinNode, N_NODES, 256, 256, EP_F32_RESID,
         hnOut, nullptr, nullptr, nullptr, hn, nullptr);

    hn = hnOut;
    hb = hbOut;
  }
}
